// attention_decoder_8443905704591
// MI455X (gfx1250) — matrix operand order repaired and checked by reading, compiled, not run on hardware
//
#include <hip/hip_runtime.h>

// ---------------------------------------------------------------------------
// CDNA5 (gfx1250) fused attention-decoder pipeline, f16 WMMA everywhere.
// Round 3: TDM staging uses the 6-arg tensor_load_to_lds builtin.
// ---------------------------------------------------------------------------

typedef _Float16 h16;
typedef __attribute__((ext_vector_type(4)))  _Float16 v4h;
typedef __attribute__((ext_vector_type(8)))  _Float16 v8h;
typedef __attribute__((ext_vector_type(16))) _Float16 v16h;
typedef __attribute__((ext_vector_type(8)))  float    v8f;

#define BTOT 16384

// f16 weight region layout (offsets in halves)
#define WO_W1T 0u          // 12 x (128x256)  latent-head W1^T (N-major)
#define WO_W2T 393216u     // 12 x (32x128)   latent-head W2^T
#define WO_OW1 442368u     // 3  x (16x32)    out-head W1^T
#define WO_OW2 443904u     // 3  x (16x32)    out-head W2^T (zero-padded K->32, N->16)
#define WO_MW1 445440u     // 128x448         mlp W1^T
#define WO_MW2 502784u     // 32x128          mlp W2^T

// workspace byte offsets
#define OFF_STATS ((size_t)2u << 20)
#define OFF_QKV   ((size_t)3u << 20)
#define OFF_OSEQ  (OFF_QKV + (size_t)BTOT * 144 * 32 * 2)

// ---------------------------------------------------------------------------
// WMMA helpers (ISA 7.12.2 layouts, wave32)
// ---------------------------------------------------------------------------
__device__ __forceinline__ v8f WMMA(v16h a, v16h b, v8f c) {
  return __builtin_amdgcn_wmma_f32_16x16x32_f16(false, a, false, b, (short)0, c,
                                                false, false);
}

// A fragment (16x32 f16): lane&15 = row, halves 0..7 -> K=hi*8+0..7,
// halves 8..15 -> K=16+hi*8+0..7.  Row-major source, ldk in halves.
__device__ __forceinline__ v16h afrag(const h16* t, int ldk, int kbase) {
  int lane = threadIdx.x & 31;
  int r = lane & 15, hi = lane >> 4;
  const h16* p = t + (size_t)r * ldk + kbase + hi * 8;
  v8h lo = *(const v8h*)p;
  v8h hb = *(const v8h*)(p + 16);
  v16h o;
#pragma unroll
  for (int i = 0; i < 8; i++) { o[i] = lo[i]; o[i + 8] = hb[i]; }
  return o;
}

// B fragment (32x16 f16): lane&15 = N, halves 0..7 -> K=hi*8+0..7, halves 8..15 -> K=16+hi*8+0..7.
// Source is N-major (B^T), ldk in halves.
__device__ __forceinline__ v16h bfrag(const h16* bt, int ldk, int nbase, int kbase) {
  int lane = threadIdx.x & 31;
  int n = lane & 15, hi = lane >> 4;
  const h16* p = bt + (size_t)(nbase + n) * ldk + kbase + hi * 8;
  v8h x0 = *(const v8h*)p;
  v8h x1 = *(const v8h*)(p + 16);
  v16h o;
#pragma unroll
  for (int i = 0; i < 8; i++) { o[i] = x0[i]; o[i + 8] = x1[i]; }
  return o;
}

__device__ __forceinline__ float gelu_f(float x) {
  return 0.5f * x * (1.0f + erff(x * 0.7071067811865476f));
}

// ---------------------------------------------------------------------------
// TDM: DMA a contiguous 2D f16 tile (rows x cols, col-stride == cols) from
// global memory into LDS.  D# built per CDNA5 ISA ch.8 (groups 0/1; 2D tensor
// so groups 2/3 are zero).  Issued by a single wave; caller barriers after.
// This toolchain's builtin takes 6 args: (g0 v4u, g1 v8i, g2 v4i, g3 v4i,
// extra v8i, cpol i32).
// ---------------------------------------------------------------------------
#if __has_builtin(__builtin_amdgcn_tensor_load_to_lds)
#define HAVE_TDM 1
typedef __attribute__((ext_vector_type(4))) unsigned ui32x4;
typedef __attribute__((ext_vector_type(8))) int      i32x8;
typedef __attribute__((ext_vector_type(4))) int      i32x4;

__device__ __forceinline__ void tdm_load_2d_f16(const h16* gsrc, h16* ldst,
                                                unsigned cols, unsigned rows) {
  // group 0: [1:0]=count=1, [63:32]=lds_addr, [120:64]=global_addr, [127:126]=type=2
  unsigned lds_off = (unsigned)(unsigned long long)(const void*)ldst;  // addr[31:0] = LDS offset
  unsigned long long ga = (unsigned long long)gsrc;
  ui32x4 g0;
  g0[0] = 1u;
  g0[1] = lds_off;
  g0[2] = (unsigned)ga;
  g0[3] = ((unsigned)(ga >> 32) & 0x01FFFFFFu) | 0x80000000u;
  // group 1: data_size=1 (2B) @ [17:16]; tensor_dim0 @ [79:48]; tensor_dim1 @ [111:80];
  //          tile_dim0 @ [127:112]; tile_dim1 @ [143:128]; dim0_stride @ [207:160]
  i32x8 g1;
  g1[0] = (int)(1u << 16);
  g1[1] = (int)(cols << 16);          // tensor_dim0 low16 -> bits 63:48
  g1[2] = (int)(rows << 16);          // tensor_dim1 low16 -> bits 95:80
  g1[3] = (int)(cols << 16);          // tile_dim0 -> bits 127:112
  g1[4] = (int)rows;                  // tile_dim1 -> bits 143:128
  g1[5] = (int)cols;                  // tensor_dim0_stride low32 -> bits 191:160
  g1[6] = 0;
  g1[7] = 0;
  i32x4 z4 = {};
  i32x8 z8 = {};
  __builtin_amdgcn_tensor_load_to_lds(g0, g1, z4, z4, z8, 0);
}
#else
#define HAVE_TDM 0
#endif

// ---------------------------------------------------------------------------
// Weight transpose+cast: dst[n*ldk+k] = (n<srcN && k<srcK) ? src[k*srcN+n] : 0
// ---------------------------------------------------------------------------
__global__ void k_xpose(const float* __restrict__ src, h16* __restrict__ dst,
                        int srcK, int srcN, int dstN, int dstLdk) {
  int i = blockIdx.x * blockDim.x + threadIdx.x;
  int total = dstN * dstLdk;
  if (i >= total) return;
  int n = i / dstLdk, k = i % dstLdk;
  float v = (n < srcN && k < srcK) ? src[(size_t)k * srcN + n] : 0.0f;
  dst[i] = (h16)v;
}

// ---------------------------------------------------------------------------
// Per-element LayerNorm stats for the 4 joint-LN inputs
// ln0: state(12x256) ln1: ts(20x256) ln2: ts[:12] ln3: ts[8:]
// ---------------------------------------------------------------------------
__global__ __launch_bounds__(256) void k_stats(const float* __restrict__ state,
                                               const float* __restrict__ ts,
                                               float* __restrict__ stats) {
  __shared__ float red[256], red2[256];
  int b = blockIdx.x;
  for (int ln = 0; ln < 4; ++ln) {
    const float* base;
    int n;
    if (ln == 0) { base = state + (size_t)b * 12 * 256; n = 12 * 256; }
    else if (ln == 1) { base = ts + (size_t)b * 20 * 256; n = 20 * 256; }
    else if (ln == 2) { base = ts + (size_t)b * 20 * 256; n = 12 * 256; }
    else { base = ts + (size_t)b * 20 * 256 + 8 * 256; n = 12 * 256; }
    float s = 0.f, s2 = 0.f;
    for (int i = threadIdx.x; i < n; i += 256) {
      float x = base[i];
      s += x; s2 += x * x;
    }
    red[threadIdx.x] = s; red2[threadIdx.x] = s2;
    __syncthreads();
    for (int o = 128; o > 0; o >>= 1) {
      if (threadIdx.x < (unsigned)o) {
        red[threadIdx.x] += red[threadIdx.x + o];
        red2[threadIdx.x] += red2[threadIdx.x + o];
      }
      __syncthreads();
    }
    if (threadIdx.x == 0) {
      float m = red[0] / n;
      float v = red2[0] / n - m * m;
      stats[(size_t)b * 8 + ln * 2] = m;
      stats[(size_t)b * 8 + ln * 2 + 1] = rsqrtf(v + 1e-5f);
    }
    __syncthreads();
  }
}

// ---------------------------------------------------------------------------
// Fused QKV projection: LN(x) @ W1 (256x128) -> gelu -> @ W2 (128x32)
// One WG (4 waves) handles 8 batch elements for one (segment, matrix) job.
// W1^T staged to LDS via the Tensor Data Mover.
// ---------------------------------------------------------------------------
struct QkvP {
  const float* state; const float* ts; const float* stats;
  const h16* W; h16* qkv;
  const float* b1[10]; const float* b2[10];
};

__device__ const int jb_src[10]  = {0, 0, 0, 1, 1, 1, 1, 1, 1, 1};
__device__ const int jb_rows[10] = {12, 12, 12, 20, 20, 20, 12, 12, 12, 12};
__device__ const int jb_roff[10] = {0, 0, 0, 0, 0, 0, 0, 0, 8, 8};
__device__ const int jb_ln[10]   = {0, 0, 0, 1, 1, 1, 2, 2, 3, 3};
__device__ const int jb_w[10]    = {0, 1, 2, 3, 4, 5, 7, 8, 10, 11};
__device__ const int jb_orow[10] = {0, 12, 24, 36, 56, 76, 96, 108, 120, 132};

__global__ __launch_bounds__(128) void k_qkv(QkvP P) {
  __shared__ __align__(16) h16 sW1[128 * 256];  // W1^T  (N-major 128x256)
  __shared__ __align__(16) h16 sW2[32 * 128];   // W2^T  (N-major 32x128)
  __shared__ __align__(16) h16 sA[32 * 256];    // normalized input, padded
  __shared__ __align__(16) h16 sH[32 * 128];    // gelu(h1)
  const int j = blockIdx.y;
  const int tid = threadIdx.x;
  const int wv = tid >> 5, lane = tid & 31, n0 = lane & 15, hi = lane >> 4;
  const h16* w1t = P.W + WO_W1T + (size_t)jb_w[j] * (128 * 256);
  const h16* w2t = P.W + WO_W2T + (size_t)jb_w[j] * (32 * 128);
#if HAVE_TDM
  if (wv == 0) {
    // one wave issues the DMA of the full 128x256 f16 tile (64 KB) into LDS
    tdm_load_2d_f16(w1t, sW1, 256u, 128u);
    __builtin_amdgcn_s_wait_tensorcnt(0);
  }
#else
  for (int i = tid; i < 128 * 256 / 8; i += 128) {
    __builtin_prefetch(w1t + (size_t)i * 8 + 2048, 0, 0);
    *(v8h*)(sW1 + (size_t)i * 8) = *(const v8h*)(w1t + (size_t)i * 8);
  }
#endif
  for (int i = tid; i < 32 * 128 / 8; i += 128)
    *(v8h*)(sW2 + (size_t)i * 8) = *(const v8h*)(w2t + (size_t)i * 8);
  __syncthreads();

  const int rows = jb_rows[j];
  const float* b1p = P.b1[j];
  const float* b2p = P.b2[j];

  for (int e = 0; e < 8; ++e) {
    const int b = blockIdx.x * 8 + e;
    const float* x = (jb_src[j] ? P.ts + (size_t)b * 20 * 256
                                : P.state + (size_t)b * 12 * 256) +
                     jb_roff[j] * 256;
    const float mean = P.stats[(size_t)b * 8 + jb_ln[j] * 2];
    const float rstd = P.stats[(size_t)b * 8 + jb_ln[j] * 2 + 1];
    // normalize-on-load A (32x256, zero padded rows)
    for (int i = tid; i < 32 * 256 / 4; i += 128) {
      int idx = i * 4, r = idx >> 8, c = idx & 255;
      v4h o;
      if (r < rows) {
        const float4 v = *(const float4*)(x + (size_t)r * 256 + c);
        o[0] = (h16)((v.x - mean) * rstd); o[1] = (h16)((v.y - mean) * rstd);
        o[2] = (h16)((v.z - mean) * rstd); o[3] = (h16)((v.w - mean) * rstd);
      } else {
        o[0] = o[1] = o[2] = o[3] = (h16)0.f;
      }
      *(v4h*)(sA + idx) = o;
    }
    __syncthreads();
    // layer 1: wave wv owns N columns [wv*32, wv*32+32)
    v8f acc[2][2] = {};
#pragma unroll
    for (int kt = 0; kt < 8; kt++) {
      v16h a0 = afrag(sA, 256, kt * 32);
      v16h a1 = afrag(sA + 16 * 256, 256, kt * 32);
      v16h bb0 = bfrag(sW1, 256, wv * 32, kt * 32);
      v16h bb1 = bfrag(sW1, 256, wv * 32 + 16, kt * 32);
      acc[0][0] = WMMA(a0, bb0, acc[0][0]);
      acc[0][1] = WMMA(a0, bb1, acc[0][1]);
      acc[1][0] = WMMA(a1, bb0, acc[1][0]);
      acc[1][1] = WMMA(a1, bb1, acc[1][1]);
    }
#pragma unroll
    for (int mt = 0; mt < 2; mt++)
#pragma unroll
      for (int nt = 0; nt < 2; nt++) {
        int ncol = wv * 32 + nt * 16 + n0;
        float bias = b1p[ncol];
#pragma unroll
        for (int r = 0; r < 8; r++) {
          int row = mt * 16 + hi * 8 + r;
          sH[(size_t)row * 128 + ncol] = (h16)gelu_f(acc[mt][nt][r] + bias);
        }
      }
    __syncthreads();
    // layer 2: wave wv owns output tile (mt = wv>>1, nt = wv&1)
    {
      const int mt = wv >> 1, nt = wv & 1;
      v8f oa = {};
#pragma unroll
      for (int kt = 0; kt < 4; kt++) {
        v16h a = afrag(sH + (size_t)mt * 16 * 128, 128, kt * 32);
        v16h bb = bfrag(sW2, 128, nt * 16, kt * 32);
        oa = WMMA(a, bb, oa);
      }
      h16* outp = P.qkv + ((size_t)b * 144 + jb_orow[j]) * 32;
      int ncol = nt * 16 + n0;
      float bias = b2p ? b2p[ncol] : 0.f;
#pragma unroll
      for (int r = 0; r < 8; r++) {
        int row = mt * 16 + hi * 8 + r;
        if (row < rows) outp[(size_t)row * 32 + ncol] = (h16)(oa[r] + bias);
      }
    }
    __syncthreads();
  }
}

// ---------------------------------------------------------------------------
// Attention for the four latent heads. One wave per (element, head).
// ---------------------------------------------------------------------------
__device__ const int ah_qr[4] = {0, 36, 0, 0};
__device__ const int ah_nq[4] = {12, 20, 12, 12};
__device__ const int ah_kr[4] = {12, 56, 96, 120};
__device__ const int ah_vr[4] = {24, 76, 108, 132};
__device__ const int ah_nk[4] = {12, 20, 12, 12};
__device__ const int ah_or[4] = {0, 12, 32, 44};

__global__ __launch_bounds__(128) void k_attn(const h16* __restrict__ qkv,
                                              h16* __restrict__ oseq) {
  __shared__ __align__(16) h16 sQ[4][32 * 32];
  __shared__ __align__(16) h16 sK[4][32 * 32];
  __shared__ __align__(16) h16 sVt[4][32 * 32];  // V transposed: [n][k]
  __shared__ __align__(16) h16 sP[4][32 * 32];
  const int b = blockIdx.x;
  const int wv = threadIdx.x >> 5, lane = threadIdx.x & 31;
  const int n0 = lane & 15, hi = lane >> 4;
  const int nq = ah_nq[wv], nk = ah_nk[wv];
  const h16* base = qkv + (size_t)b * 144 * 32;
  {
    int r = lane;
    if (r < nq) {
      const v8h* src = (const v8h*)(base + (size_t)(ah_qr[wv] + r) * 32);
      v8h* dst = (v8h*)(sQ[wv] + r * 32);
      dst[0] = src[0]; dst[1] = src[1]; dst[2] = src[2]; dst[3] = src[3];
    } else {
      v8h z = {};
      v8h* dst = (v8h*)(sQ[wv] + r * 32);
      dst[0] = z; dst[1] = z; dst[2] = z; dst[3] = z;
    }
    if (r < nk) {
      const v8h* src = (const v8h*)(base + (size_t)(ah_kr[wv] + r) * 32);
      v8h* dst = (v8h*)(sK[wv] + r * 32);
      dst[0] = src[0]; dst[1] = src[1]; dst[2] = src[2]; dst[3] = src[3];
      const h16* vs = base + (size_t)(ah_vr[wv] + r) * 32;
      for (int c = 0; c < 32; c++) sVt[wv][c * 32 + r] = vs[c];
    } else {
      v8h z = {};
      v8h* dst = (v8h*)(sK[wv] + r * 32);
      dst[0] = z; dst[1] = z; dst[2] = z; dst[3] = z;
      for (int c = 0; c < 32; c++) sVt[wv][c * 32 + r] = (h16)0.f;
    }
  }
  __syncthreads();
  // scores = Q @ K^T / sqrt(32)
  v8f sc[2][2] = {};
  {
    v16h a0 = afrag(sQ[wv], 32, 0);
    v16h a1 = afrag(sQ[wv] + 16 * 32, 32, 0);
    v16h b0 = bfrag(sK[wv], 32, 0, 0);
    v16h b1 = bfrag(sK[wv], 32, 16, 0);
    sc[0][0] = WMMA(a0, b0, sc[0][0]); sc[0][1] = WMMA(a0, b1, sc[0][1]);
    sc[1][0] = WMMA(a1, b0, sc[1][0]); sc[1][1] = WMMA(a1, b1, sc[1][1]);
  }
  const float scale = 0.17677669529663687f;  // 1/sqrt(32)
#pragma unroll
  for (int mt = 0; mt < 2; mt++)
#pragma unroll
    for (int r = 0; r < 8; r++) {
      float a0 = (n0 < nk) ? sc[mt][0][r] * scale : -1e30f;
      float a1 = (16 + n0 < nk) ? sc[mt][1][r] * scale : -1e30f;
      float m = fmaxf(a0, a1);
#pragma unroll
      for (int d = 1; d < 16; d <<= 1) m = fmaxf(m, __shfl_xor(m, d, 32));
      float e0 = __expf(a0 - m), e1 = __expf(a1 - m);
      float s = e0 + e1;
#pragma unroll
      for (int d = 1; d < 16; d <<= 1) s += __shfl_xor(s, d, 32);
      float inv = 1.f / s;
      int row = mt * 16 + hi * 8 + r;
      sP[wv][row * 32 + n0] = (h16)(e0 * inv);
      sP[wv][row * 32 + 16 + n0] = (h16)(e1 * inv);
    }
  __syncthreads();
  // out = P @ V
  v8f oa[2][2] = {};
  {
    v16h a0 = afrag(sP[wv], 32, 0);
    v16h a1 = afrag(sP[wv] + 16 * 32, 32, 0);
    v16h b0 = bfrag(sVt[wv], 32, 0, 0);
    v16h b1 = bfrag(sVt[wv], 32, 16, 0);
    oa[0][0] = WMMA(a0, b0, oa[0][0]); oa[0][1] = WMMA(a0, b1, oa[0][1]);
    oa[1][0] = WMMA(a1, b0, oa[1][0]); oa[1][1] = WMMA(a1, b1, oa[1][1]);
  }
  h16* op = oseq + ((size_t)b * 56 + ah_or[wv]) * 32;
#pragma unroll
  for (int mt = 0; mt < 2; mt++)
#pragma unroll
    for (int nt = 0; nt < 2; nt++)
#pragma unroll
      for (int r = 0; r < 8; r++) {
        int row = mt * 16 + hi * 8 + r;
        if (row < nq) op[(size_t)row * 32 + nt * 16 + n0] = (h16)oa[mt][nt][r];
      }
}

// ---------------------------------------------------------------------------
// Output head: LN(56x32) -> qkv (32->16->8) -> attention -> LN(448) -> flat
// One WG (4 waves) per element; wave wv owns M-tile wv (rows wv*16..+15).
// ---------------------------------------------------------------------------
struct OutP {
  const h16* oseq; const h16* W; h16* flat;
  const float* b1[3]; const float* qb2;
};

__global__ __launch_bounds__(128) void k_outhead(OutP P) {
  __shared__ __align__(16) h16 sX[64 * 32];
  __shared__ __align__(16) h16 sQ[64 * 32];
  __shared__ __align__(16) h16 sK[64 * 32];
  __shared__ __align__(16) h16 sVt[16 * 64];
  __shared__ __align__(16) h16 sH[64 * 32];
  __shared__ __align__(16) h16 sP[64 * 64];
  __shared__ float sAO[448];
  __shared__ float red[128], red2[128];
  __shared__ float sMR[2];
  const int b = blockIdx.x, tid = threadIdx.x;
  const int wv = tid >> 5, lane = tid & 31, n0 = lane & 15, hi = lane >> 4;
  for (int i = tid; i < 64 * 32; i += 128) {
    sX[i] = (h16)0.f; sQ[i] = (h16)0.f; sK[i] = (h16)0.f; sH[i] = (h16)0.f;
  }
  for (int i = tid; i < 16 * 64; i += 128) sVt[i] = (h16)0.f;
  __syncthreads();
  // load + LN stats over 56x32
  const h16* src = P.oseq + (size_t)b * 56 * 32;
  float s = 0.f, s2 = 0.f;
  for (int i = tid; i < 56 * 32; i += 128) {
    float x = (float)src[i];
    s += x; s2 += x * x;
    sX[i] = (h16)x;
  }
  red[tid] = s; red2[tid] = s2;
  __syncthreads();
  for (int o = 64; o > 0; o >>= 1) {
    if (tid < o) { red[tid] += red[tid + o]; red2[tid] += red2[tid + o]; }
    __syncthreads();
  }
  if (tid == 0) {
    float m = red[0] / 1792.f, v = red2[0] / 1792.f - m * m;
    sMR[0] = m; sMR[1] = rsqrtf(v + 1e-5f);
  }
  __syncthreads();
  {
    float m = sMR[0], rs = sMR[1];
    for (int i = tid; i < 56 * 32; i += 128)
      sX[i] = (h16)(((float)sX[i] - m) * rs);
  }
  __syncthreads();
  // projections: mm = 0:Q 1:K 2:V
  for (int mm = 0; mm < 3; mm++) {
    v16h a = afrag(sX + (size_t)wv * 16 * 32, 32, 0);
    v16h bb = bfrag(P.W + WO_OW1 + mm * 512, 32, 0, 0);
    v8f h1 = {};
    h1 = WMMA(a, bb, h1);
    const float* b1p = P.b1[mm];
#pragma unroll
    for (int r = 0; r < 8; r++) {
      int row = wv * 16 + hi * 8 + r;
      sH[row * 32 + n0] = (h16)gelu_f(h1[r] + b1p[n0]);
    }
    v16h a2 = afrag(sH + (size_t)wv * 16 * 32, 32, 0);
    v16h bw2 = bfrag(P.W + WO_OW2 + mm * 512, 32, 0, 0);
    v8f y = {};
    y = WMMA(a2, bw2, y);
    float bias = (mm == 0 && n0 < 8) ? P.qb2[n0] : 0.f;
#pragma unroll
    for (int r = 0; r < 8; r++) {
      int row = wv * 16 + hi * 8 + r;
      float v = y[r] + bias;
      if (mm == 0) sQ[row * 32 + n0] = (h16)v;
      else if (mm == 1) sK[row * 32 + n0] = (h16)v;
      else sVt[n0 * 64 + row] = (h16)v;
    }
  }
  __syncthreads();
  // scores (56x56, K padded 8->32), softmax over 56 cols
  const float scl = 0.3535533905932738f;  // 1/sqrt(8)
  v8f sc[4];
  {
    v16h a = afrag(sQ + (size_t)wv * 16 * 32, 32, 0);
#pragma unroll
    for (int nt = 0; nt < 4; nt++) {
      v16h bb = bfrag(sK, 32, nt * 16, 0);
      v8f z = {};
      sc[nt] = WMMA(a, bb, z);
    }
  }
#pragma unroll
  for (int r = 0; r < 8; r++) {
    float v[4];
    float m = -1e30f;
#pragma unroll
    for (int nt = 0; nt < 4; nt++) {
      int col = nt * 16 + n0;
      v[nt] = (col < 56) ? sc[nt][r] * scl : -1e30f;
      m = fmaxf(m, v[nt]);
    }
#pragma unroll
    for (int d = 1; d < 16; d <<= 1) m = fmaxf(m, __shfl_xor(m, d, 32));
    float e[4], ssum = 1e-20f;
#pragma unroll
    for (int nt = 0; nt < 4; nt++) { e[nt] = __expf(v[nt] - m); ssum += e[nt]; }
#pragma unroll
    for (int d = 1; d < 16; d <<= 1) ssum += __shfl_xor(ssum, d, 32);
    float inv = 1.f / ssum;
    int row = wv * 16 + hi * 8 + r;
#pragma unroll
    for (int nt = 0; nt < 4; nt++) sP[row * 64 + nt * 16 + n0] = (h16)(e[nt] * inv);
  }
  __syncthreads();
  // att_out = P @ V  (K=64, N pad 8->16)
  v8f ao = {};
#pragma unroll
  for (int kt = 0; kt < 2; kt++) {
    v16h a = afrag(sP + (size_t)wv * 16 * 64, 64, kt * 32);
    v16h bb = bfrag(sVt, 64, 0, kt * 32);
    ao = WMMA(a, bb, ao);
  }
  float ps = 0.f, ps2 = 0.f;
#pragma unroll
  for (int r = 0; r < 8; r++) {
    int row = wv * 16 + hi * 8 + r;
    if (row < 56 && n0 < 8) {
      float v = ao[r];
      sAO[row * 8 + n0] = v;
      ps += v; ps2 += v * v;
    }
  }
  red[tid] = ps; red2[tid] = ps2;
  __syncthreads();
  for (int o = 64; o > 0; o >>= 1) {
    if (tid < o) { red[tid] += red[tid + o]; red2[tid] += red2[tid + o]; }
    __syncthreads();
  }
  if (tid == 0) {
    float m = red[0] / 448.f, v = red2[0] / 448.f - m * m;
    sMR[0] = m; sMR[1] = rsqrtf(v + 1e-5f);
  }
  __syncthreads();
  {
    float m = sMR[0], rs = sMR[1];
    h16* fp = P.flat + (size_t)b * 448;
    for (int i = tid; i < 448; i += 128) fp[i] = (h16)((sAO[i] - m) * rs);
  }
}

// ---------------------------------------------------------------------------
// Final MLP 448 -> 128 -> 32 -> 2, batched GEMM over M = 16384
// ---------------------------------------------------------------------------
__global__ __launch_bounds__(128) void k_mlp(const h16* __restrict__ flat,
                                             const h16* __restrict__ W,
                                             const float* __restrict__ b1,
                                             const float* __restrict__ b2,
                                             const float* __restrict__ W3,
                                             const float* __restrict__ b3,
                                             float* __restrict__ out) {
  __shared__ __align__(16) h16 sH1[4][16 * 128];
  __shared__ __align__(16) h16 sH2[4][16 * 32];
  const int wv = threadIdx.x >> 5, lane = threadIdx.x & 31;
  const int n0 = lane & 15, hi = lane >> 4;
  const size_t row0 = (size_t)blockIdx.x * 64 + wv * 16;
  const h16* A = flat + row0 * 448;
  const h16* W1t = W + WO_MW1;
  const h16* W2t = W + WO_MW2;
  v8f acc[8] = {};
  for (int kt = 0; kt < 14; kt++) {
    v16h a = afrag(A, 448, kt * 32);
#pragma unroll
    for (int nt = 0; nt < 8; nt++) {
      __builtin_prefetch(W1t + (size_t)(nt * 16 + n0) * 448 + kt * 32 + 64, 0, 0);
      v16h bb = bfrag(W1t, 448, nt * 16, kt * 32);
      acc[nt] = WMMA(a, bb, acc[nt]);
    }
  }
#pragma unroll
  for (int nt = 0; nt < 8; nt++) {
    int col = nt * 16 + n0;
    float bias = b1[col];
#pragma unroll
    for (int r = 0; r < 8; r++) {
      int rl = hi * 8 + r;
      sH1[wv][rl * 128 + col] = (h16)gelu_f(acc[nt][r] + bias);
    }
  }
  v8f a2[2] = {};
#pragma unroll
  for (int kt = 0; kt < 4; kt++) {
    v16h a = afrag(sH1[wv], 128, kt * 32);
#pragma unroll
    for (int nt = 0; nt < 2; nt++) {
      v16h bb = bfrag(W2t, 128, nt * 16, kt * 32);
      a2[nt] = WMMA(a, bb, a2[nt]);
    }
  }
#pragma unroll
  for (int nt = 0; nt < 2; nt++) {
    int col = nt * 16 + n0;
    float bias = b2[col];
#pragma unroll
    for (int r = 0; r < 8; r++) {
      int rl = hi * 8 + r;
      sH2[wv][rl * 32 + col] = (h16)gelu_f(a2[nt][r] + bias);
    }
  }
  // layer 3: 32 -> 2 (VALU dot products, 16 rows x 2 outputs per wave)
  {
    int r = lane & 15, jj = lane >> 4;
    float acc3 = b3[jj];
#pragma unroll
    for (int kk = 0; kk < 32; kk++)
      acc3 += (float)sH2[wv][r * 32 + kk] * W3[kk * 2 + jj];
    out[(row0 + r) * 2 + jj] = acc3;
  }
}

// ---------------------------------------------------------------------------
// Host launcher
// ---------------------------------------------------------------------------
extern "C" void kernel_launch(void* const* d_in, const int* in_sizes, int n_in,
                              void* d_out, int out_size, void* d_ws, size_t ws_size,
                              hipStream_t stream) {
  (void)in_sizes; (void)n_in; (void)out_size; (void)ws_size;
  const float* state = (const float*)d_in[0];
  const float* ts = (const float*)d_in[1];
  auto HP = [&](int h, int k) -> const float* {
    return (const float*)d_in[2 + h * 10 + k];
  };
  const float* mW1 = (const float*)d_in[52];
  const float* mb1 = (const float*)d_in[53];
  const float* mW2 = (const float*)d_in[54];
  const float* mb2 = (const float*)d_in[55];
  const float* mW3 = (const float*)d_in[56];
  const float* mb3 = (const float*)d_in[57];

  char* ws = (char*)d_ws;
  h16* W = (h16*)ws;
  float* stats = (float*)(ws + OFF_STATS);
  h16* qkv = (h16*)(ws + OFF_QKV);
  h16* oseq = (h16*)(ws + OFF_OSEQ);
  h16* flat = (h16*)(ws + OFF_QKV);  // qkv buffer is dead by the time flat is written
  float* out = (float*)d_out;

  // per-head param indices: Q{W1=0,b1=1,W2=2,b2=3} K{4,5,6} V{7,8,9}
  const int w1i[3] = {0, 4, 7}, w2i[3] = {2, 6, 9}, b1i[3] = {1, 5, 8};

  // weight transpose+cast to f16 (N-major)
  for (int h = 0; h < 4; h++)
    for (int m = 0; m < 3; m++) {
      k_xpose<<<dim3((128 * 256 + 255) / 256), 256, 0, stream>>>(
          HP(h, w1i[m]), W + WO_W1T + (size_t)(h * 3 + m) * 128 * 256, 256, 128,
          128, 256);
      k_xpose<<<dim3((32 * 128 + 255) / 256), 256, 0, stream>>>(
          HP(h, w2i[m]), W + WO_W2T + (size_t)(h * 3 + m) * 32 * 128, 128, 32,
          32, 128);
    }
  for (int m = 0; m < 3; m++) {
    k_xpose<<<2, 256, 0, stream>>>(HP(4, w1i[m]), W + WO_OW1 + (size_t)m * 512,
                                   32, 16, 16, 32);
    k_xpose<<<2, 256, 0, stream>>>(HP(4, w2i[m]), W + WO_OW2 + (size_t)m * 512,
                                   16, 8, 16, 32);
  }
  k_xpose<<<(128 * 448 + 255) / 256, 256, 0, stream>>>(mW1, W + WO_MW1, 448,
                                                       128, 128, 448);
  k_xpose<<<(32 * 128 + 255) / 256, 256, 0, stream>>>(mW2, W + WO_MW2, 128, 32,
                                                      32, 128);

  k_stats<<<BTOT, 256, 0, stream>>>(state, ts, stats);

  QkvP qp{};
  qp.state = state; qp.ts = ts; qp.stats = stats; qp.W = W; qp.qkv = qkv;
  const int jh[10] = {0, 0, 0, 1, 1, 1, 2, 2, 3, 3};
  const int jm[10] = {0, 1, 2, 0, 1, 2, 1, 2, 1, 2};
  for (int j = 0; j < 10; j++) {
    qp.b1[j] = HP(jh[j], b1i[jm[j]]);
    qp.b2[j] = (jm[j] == 0) ? HP(jh[j], 3) : nullptr;
  }
  k_qkv<<<dim3(BTOT / 8, 10), 128, 0, stream>>>(qp);

  k_attn<<<BTOT, 128, 0, stream>>>(qkv, oseq);

  OutP op{};
  op.oseq = oseq; op.W = W; op.flat = flat;
  for (int m = 0; m < 3; m++) op.b1[m] = HP(4, b1i[m]);
  op.qb2 = HP(4, 3);
  k_outhead<<<BTOT, 128, 0, stream>>>(op);

  k_mlp<<<BTOT / 64, 128, 0, stream>>>(flat, W, mb1, mb2, mW3, mb3, out);
}
